// DSAttention_75213467287636
// MI455X (gfx1250) — matrix operand order repaired and checked by reading, not compiled, not run on hardware
//
#include <hip/hip_runtime.h>

// ---------------------------------------------------------------------------
// Causal attention (DSAttention):  V = softmax(scale*(tau*Q.K^T + delta)) @ V
// B=2, L=S=2048, H=8, E=64, fp32 in/out.
// Flash-attention, 3xBF16 fp32-emulation on v_wmma_f32_16x16x32_bf16.
// - K/V converted to packed (hi,lo) bf16 pairs cooperatively at LDS-load time
//   so WMMA operand builds are pure ds_load_b128 (zero VALU).
// - Softmax row-sums computed as P x ones via WMMA (no shuffle network).
// - Raw v_exp_f32 (2^x) for softmax exponentials.
// ---------------------------------------------------------------------------

typedef __attribute__((ext_vector_type(16))) __bf16 v16bf;
typedef __attribute__((ext_vector_type(8)))  float  v8f;

#define B_DIM 2
#define L_DIM 2048
#define H_DIM 8
#define E_DIM 64
#define S_DIM 2048
#define QBLK  64
#define ROWS  512               // H_DIM*E_DIM: row stride of Q/K/V in floats
#define WPITCH 36               // dwords per packed row: 16B-aligned, bank-perm
#define QPITCH 65               // f32 Q staging pitch
#define LOG2E 1.4426950408889634f

#if __has_builtin(__builtin_amdgcn_exp2f)
#define EXP2F(x) __builtin_amdgcn_exp2f(x)
#else
#define EXP2F(x) exp2f(x)
#endif

__device__ __forceinline__ __bf16 bhi(float x) { return (__bf16)x; }
__device__ __forceinline__ __bf16 blo(float x) {
    float h = (float)((__bf16)x);
    return (__bf16)(x - h);
}
__device__ __forceinline__ unsigned short bfbits(__bf16 v) {
    return __builtin_bit_cast(unsigned short, v);
}
// pack two bf16(hi parts) of (a,b): low half = a
__device__ __forceinline__ unsigned packhi(float a, float b) {
    return (unsigned)bfbits(bhi(a)) | ((unsigned)bfbits(bhi(b)) << 16);
}
__device__ __forceinline__ unsigned packlo(float a, float b) {
    return (unsigned)bfbits(blo(a)) | ((unsigned)bfbits(blo(b)) << 16);
}

// A-matrix (16x32 bf16) K-index (ISA 7.12.2): lanes0-15 K {0..7,16..23},
// lanes16-31 K {8..15,24..31}.
__device__ __forceinline__ int a_kidx(int idx, int hg) {
    int v = idx >> 1, pos = idx & 1;
    return (v < 4) ? (2 * v + pos + 8 * hg) : (16 + 2 * (v - 4) + pos + 8 * hg);
}

__device__ __forceinline__ v8f wmma_bf16(v16bf a, v16bf b, v8f c) {
    return __builtin_amdgcn_wmma_f32_16x16x32_bf16(false, a, false, b,
                                                   (short)0, c, false, false);
}

// Load a v16bf operand as two 16B LDS reads (ds_load_b128), zero VALU.
__device__ __forceinline__ v16bf ld_op2(const unsigned* p0, const unsigned* p1) {
    union { uint4 u[2]; v16bf v; } x;
    x.u[0] = *reinterpret_cast<const uint4*>(p0);
    x.u[1] = *reinterpret_cast<const uint4*>(p1);
    return x.v;
}

__global__ __launch_bounds__(128, 1)
void dsattn_fa_kernel(const float* __restrict__ Qg,
                      const float* __restrict__ Kg,
                      const float* __restrict__ Vg,
                      const float* __restrict__ taug,
                      const float* __restrict__ deltag,
                      float* __restrict__ outg) {
    // LDS (dwords): khi|klo|vhi|vlo tiles (64x36 each) + per-wave P staging
    // (overlapped with the one-shot f32 Q staging buffer).
    __shared__ unsigned smu[4 * 64 * WPITCH + 4 * 2 * 16 * WPITCH];
    unsigned* khi = smu;
    unsigned* klo = smu + 1 * 64 * WPITCH;
    unsigned* vhi = smu + 2 * 64 * WPITCH;
    unsigned* vlo = smu + 3 * 64 * WPITCH;
    unsigned* preg_base = smu + 4 * 64 * WPITCH;
    float*    qstage = reinterpret_cast<float*>(preg_base);   // 64x65 f32 fits

    const int tid  = threadIdx.x;
    const int lane = tid & 31;
    const int wv   = tid >> 5;
    const int r16  = lane & 15;
    const int hg   = lane >> 4;

    const int bid = blockIdx.x;                 // (b*H + h)*32 + qb
    const int qb  = bid & 31;
    const int bh  = bid >> 5;
    const int h   = bh & (H_DIM - 1);
    const int b   = bh >> 3;
    const int q0  = qb * QBLK;

    const float* qbase = Qg + (size_t)b * L_DIM * ROWS + h * E_DIM;
    const float* kbase = Kg + (size_t)b * S_DIM * ROWS + h * E_DIM;
    const float* vbase = Vg + (size_t)b * S_DIM * ROWS + h * E_DIM;

    const float scale2 = 0.125f * LOG2E;        // fold 1/sqrt(E) and log2(e)
    const float alpha  = taug[b] * scale2;

    // per-wave P staging: hi shorts [16][72], lo shorts [16][72]
    unsigned*       pu  = preg_base + wv * (2 * 16 * WPITCH);
    unsigned short* phs = reinterpret_cast<unsigned short*>(pu);
    unsigned short* pls = phs + 16 * 2 * WPITCH;
    unsigned*       pul = pu + 16 * WPITCH;

    // constant all-ones B operand for row-sum WMMA
    v16bf onesb;
    #pragma unroll
    for (int i = 0; i < 16; ++i) onesb[i] = (__bf16)1.0f;

    // ---- stage Q (f32), build hi/lo bf16 A-operands once --------------------
    #pragma unroll
    for (int i = 0; i < 8; ++i) {
        int idx = tid + i * 128;
        int r = idx >> 4, c4 = (idx & 15) << 2;
        const float4 g = *reinterpret_cast<const float4*>(
            qbase + (size_t)(q0 + r) * ROWS + c4);
        float* d = qstage + r * QPITCH + c4;
        d[0] = g.x; d[1] = g.y; d[2] = g.z; d[3] = g.w;
    }
    __syncthreads();

    v16bf qhi[2], qlo[2];
    #pragma unroll
    for (int c = 0; c < 2; ++c)
        #pragma unroll
        for (int idx = 0; idx < 16; ++idx) {
            int e   = c * 32 + a_kidx(idx, hg);
            float x = qstage[(wv * 16 + r16) * QPITCH + e];
            qhi[c][idx] = bhi(x);
            qlo[c][idx] = blo(x);
        }
    __syncthreads();    // Q staging region becomes P staging after this

    // ---- online-softmax state ----------------------------------------------
    float m_run[8], l_run[8];
    v8f   O[4];
    #pragma unroll
    for (int j = 0; j < 8; ++j) { m_run[j] = -1e30f; l_run[j] = 0.0f; }
    #pragma unroll
    for (int dt = 0; dt < 4; ++dt)
        #pragma unroll
        for (int j = 0; j < 8; ++j) O[dt][j] = 0.0f;

    const int nblk = qb + 1;
    for (int sb = 0; sb < nblk; ++sb) {
        const int s0 = sb * QBLK;

        // ---- cooperative load + hi/lo bf16 pair-pack: K (rows=key n) -------
        #pragma unroll
        for (int i = 0; i < 8; ++i) {
            int idx = tid + i * 128;
            int n = idx >> 4, c4 = (idx & 15) << 2;
            const float4 g = *reinterpret_cast<const float4*>(
                kbase + (size_t)(s0 + n) * ROWS + c4);
            int w = n * WPITCH + (c4 >> 1);
            khi[w]     = packhi(g.x, g.y);
            khi[w + 1] = packhi(g.z, g.w);
            klo[w]     = packlo(g.x, g.y);
            klo[w + 1] = packlo(g.z, g.w);
        }
        // ---- V transposed: vhi[d][s/2] pairs along s -----------------------
        #pragma unroll
        for (int i = 0; i < 4; ++i) {
            int idx = tid + i * 128;
            int sp = idx >> 4, d0 = (idx & 15) << 2;
            int s  = sp * 2;
            const float4 r0 = *reinterpret_cast<const float4*>(
                vbase + (size_t)(s0 + s) * ROWS + d0);
            const float4 r1 = *reinterpret_cast<const float4*>(
                vbase + (size_t)(s0 + s + 1) * ROWS + d0);
            vhi[(d0 + 0) * WPITCH + sp] = packhi(r0.x, r1.x);
            vhi[(d0 + 1) * WPITCH + sp] = packhi(r0.y, r1.y);
            vhi[(d0 + 2) * WPITCH + sp] = packhi(r0.z, r1.z);
            vhi[(d0 + 3) * WPITCH + sp] = packhi(r0.w, r1.w);
            vlo[(d0 + 0) * WPITCH + sp] = packlo(r0.x, r1.x);
            vlo[(d0 + 1) * WPITCH + sp] = packlo(r0.y, r1.y);
            vlo[(d0 + 2) * WPITCH + sp] = packlo(r0.z, r1.z);
            vlo[(d0 + 3) * WPITCH + sp] = packlo(r0.w, r1.w);
        }

        // delta (pre-scaled to log2 domain) for my 4 columns
        float dl[4];
        #pragma unroll
        for (int t = 0; t < 4; ++t)
            dl[t] = deltag[(size_t)b * S_DIM + s0 + t * 16 + r16] * scale2;

        __syncthreads();

        // ---- S = Q.K^T (3xBF16), 4 tiles; operands = pure ds_load_b128 -----
        v8f st[4];
        #pragma unroll
        for (int t = 0; t < 4; ++t) {
            v8f acc;
            #pragma unroll
            for (int j = 0; j < 8; ++j) acc[j] = 0.0f;
            #pragma unroll
            for (int c = 0; c < 2; ++c) {
                const int wbase = (t * 16 + r16) * WPITCH + c * 16 + 4 * hg;
                v16bf khb = ld_op2(khi + wbase, khi + wbase + 8);
                v16bf klb = ld_op2(klo + wbase, klo + wbase + 8);
                acc = wmma_bf16(qhi[c], khb, acc);
                acc = wmma_bf16(qhi[c], klb, acc);
                acc = wmma_bf16(qlo[c], khb, acc);
            }
            #pragma unroll
            for (int j = 0; j < 8; ++j)
                st[t][j] = alpha * acc[j] + dl[t];
        }
        if (sb == qb) {         // causal mask only on the diagonal block
            #pragma unroll
            for (int t = 0; t < 4; ++t) {
                const int col = s0 + t * 16 + r16;
                #pragma unroll
                for (int j = 0; j < 8; ++j) {
                    int row_g = q0 + wv * 16 + j + 8 * hg;
                    if (col > row_g) st[t][j] = -1e30f;
                }
            }
        }

        // ---- online softmax (base-2): row max + rescale --------------------
        float mnew[8], corr[8];
        #pragma unroll
        for (int j = 0; j < 8; ++j) {
            float v = fmaxf(fmaxf(st[0][j], st[1][j]), fmaxf(st[2][j], st[3][j]));
            #pragma unroll
            for (int mb = 1; mb < 16; mb <<= 1)
                v = fmaxf(v, __shfl_xor(v, mb, 32));
            mnew[j] = fmaxf(m_run[j], v);
            corr[j] = EXP2F(m_run[j] - mnew[j]);
            m_run[j] = mnew[j];
        }
        #pragma unroll
        for (int t = 0; t < 4; ++t)
            #pragma unroll
            for (int j = 0; j < 8; ++j)
                st[t][j] = EXP2F(st[t][j] - mnew[j]);
        #pragma unroll
        for (int dt = 0; dt < 4; ++dt)
            #pragma unroll
            for (int j = 0; j < 8; ++j)
                O[dt][j] *= corr[j];

        // ---- stage P: hi/lo b16 stores into pair-layout --------------------
        #pragma unroll
        for (int t = 0; t < 4; ++t)
            #pragma unroll
            for (int j = 0; j < 8; ++j) {
                float p  = st[t][j];
                int rloc = j + 8 * hg;
                int sidx = rloc * (2 * WPITCH) + t * 16 + r16;
                phs[sidx] = bfbits(bhi(p));
                pls[sidx] = bfbits(blo(p));
            }
        asm volatile("s_wait_dscnt 0" ::: "memory");
        __builtin_amdgcn_wave_barrier();

        // ---- P A-operands: pure b128 loads ---------------------------------
        v16bf phiv[2], plov[2];
        #pragma unroll
        for (int c = 0; c < 2; ++c) {
            const int wb = r16 * WPITCH + c * 16 + 4 * hg;
            phiv[c] = ld_op2(pu  + wb, pu  + wb + 8);
            plov[c] = ld_op2(pul + wb, pul + wb + 8);
        }

        // ---- row-sum l via WMMA: (Phi+Plo) x ones --------------------------
        {
            v8f rsum;
            #pragma unroll
            for (int j = 0; j < 8; ++j) rsum[j] = 0.0f;
            rsum = wmma_bf16(phiv[0], onesb, rsum);
            rsum = wmma_bf16(plov[0], onesb, rsum);
            rsum = wmma_bf16(phiv[1], onesb, rsum);
            rsum = wmma_bf16(plov[1], onesb, rsum);
            #pragma unroll
            for (int j = 0; j < 8; ++j)
                l_run[j] = l_run[j] * corr[j] + rsum[j];
        }

        // ---- O += P.V (3xBF16) ---------------------------------------------
        #pragma unroll
        for (int dt = 0; dt < 4; ++dt) {
            v8f acc = O[dt];
            #pragma unroll
            for (int c = 0; c < 2; ++c) {
                const int wbase = (dt * 16 + r16) * WPITCH + c * 16 + 4 * hg;
                v16bf vhb = ld_op2(vhi + wbase, vhi + wbase + 8);
                v16bf vlb = ld_op2(vlo + wbase, vlo + wbase + 8);
                acc = wmma_bf16(phiv[c], vhb, acc);
                acc = wmma_bf16(phiv[c], vlb, acc);
                acc = wmma_bf16(plov[c], vhb, acc);
            }
            O[dt] = acc;
        }

        __syncthreads();
    }

    // ---- epilogue: O / l -> out (B, L, H, E) -------------------------------
    float linv[8];
    #pragma unroll
    for (int j = 0; j < 8; ++j) linv[j] = 1.0f / l_run[j];

    #pragma unroll
    for (int dt = 0; dt < 4; ++dt)
        #pragma unroll
        for (int j = 0; j < 8; ++j) {
            int rloc  = j + 8 * hg;
            int row_g = q0 + wv * 16 + rloc;
            int d     = dt * 16 + r16;
            outg[((size_t)b * L_DIM + row_g) * ROWS + h * E_DIM + d] =
                O[dt][j] * linv[j];
        }
}

extern "C" void kernel_launch(void* const* d_in, const int* in_sizes, int n_in,
                              void* d_out, int out_size, void* d_ws, size_t ws_size,
                              hipStream_t stream) {
    (void)in_sizes; (void)n_in; (void)d_ws; (void)ws_size; (void)out_size;
    const float* Q     = (const float*)d_in[0];
    const float* K     = (const float*)d_in[1];
    const float* V     = (const float*)d_in[2];
    const float* tau   = (const float*)d_in[3];
    const float* delta = (const float*)d_in[4];
    float* out = (float*)d_out;

    dim3 grid(B_DIM * H_DIM * (L_DIM / QBLK));   // 512 workgroups
    dim3 block(128);                             // 4 waves (wave32)
    dsattn_fa_kernel<<<grid, block, 0, stream>>>(Q, K, V, tau, delta, out);
}
